// GraphAttentionLayer_61048665145352
// MI455X (gfx1250) — matrix operand order repaired and checked by reading, not compiled, not run on hardware
//
#include <hip/hip_runtime.h>

typedef __attribute__((ext_vector_type(16))) _Float16 v16h;
typedef __attribute__((ext_vector_type(8)))  float    v8f;
typedef __attribute__((ext_vector_type(4)))  float    f4;

#define N_NODES 8192
#define F_IN    128
#define F_OUT   64

// ---------------------------------------------------------------------------
// Kernel A: h = x @ W   (8192x128 @ 128x64). Also emits hT in f16 (64 x 8192)
// for the WMMA B-operand of the fused attention kernel.
// ---------------------------------------------------------------------------
__global__ void gat_h_kernel(const float* __restrict__ x,
                             const float* __restrict__ W,
                             float* __restrict__ h32,
                             _Float16* __restrict__ hT) {
    int gid = blockIdx.x * blockDim.x + threadIdx.x;   // 0 .. 8192*64-1
    int i = gid >> 6;          // node
    int n = gid & 63;          // output feature
    const float* xr = x + (size_t)i * F_IN;
    float acc = 0.0f;
#pragma unroll 8
    for (int k = 0; k < F_IN; ++k)
        acc += xr[k] * W[k * F_OUT + n];
    h32[(size_t)i * F_OUT + n] = acc;
    hT[(size_t)n * N_NODES + i] = (_Float16)acc;       // transposed f16 copy
}

// ---------------------------------------------------------------------------
// Kernel B: s1 = h @ a[:64], s2 = h @ a[64:]
// ---------------------------------------------------------------------------
__global__ void gat_s_kernel(const float* __restrict__ h32,
                             const float* __restrict__ a,
                             float* __restrict__ s1,
                             float* __restrict__ s2) {
    int i = blockIdx.x * blockDim.x + threadIdx.x;     // 0 .. 8191
    const float* hr = h32 + (size_t)i * F_OUT;
    float a1 = 0.0f, a2 = 0.0f;
#pragma unroll
    for (int n = 0; n < F_OUT; ++n) {
        float hv = hr[n];
        a1 += hv * a[n];
        a2 += hv * a[F_OUT + n];
    }
    s1[i] = a1;
    s2[i] = a2;
}

// ---------------------------------------------------------------------------
// Main fused kernel: per 16-row tile, stream adj (non-temporal), build
// p = adj>0 ? exp(leaky_relu(s1+s2)) : 0 as the WMMA A operand (f16),
// accumulate P(16xK) x H(Kx64) with v_wmma_f32_16x16x32_f16 (f32 acc),
// track exact f32 row sums, then normalize + ELU.
// 128 threads/block = 4 waves; waves split the K (j) dimension interleaved.
// ---------------------------------------------------------------------------
__global__ void __launch_bounds__(128)
gat_main_kernel(const float* __restrict__ adj,
                const float* __restrict__ s1g,
                const float* __restrict__ s2g,
                const _Float16* __restrict__ hT,
                float* __restrict__ out) {
    __shared__ float cbuf[4][16][64];   // per-wave partial C tiles
    __shared__ float rbuf[4][32];       // per-wave per-lane partial row sums

    const int row0 = blockIdx.x * 16;
    const int wave = threadIdx.x >> 5;
    const int lane = threadIdx.x & 31;

    // A-operand k order chosen here (contiguous, matches B): lane m = lane&15 is matrix row M,
    // koff = (lane>>4)*16; element e -> K = koff+e.
    const int m    = lane & 15;
    const int koff = (lane >> 4) * 16;
    const float s1v = s1g[row0 + m];

    v8f c0 = {}, c1 = {}, c2 = {}, c3 = {};
    float rs0 = 0.0f, rs1 = 0.0f, rs2 = 0.0f, rs3 = 0.0f;

    const float* ap = adj + (size_t)(row0 + m) * N_NODES + (wave * 32) + koff;
    const float* sp = s2g + (wave * 32) + koff;
    // B-operand k order chosen here (contiguous, matches A): lane holds column N = lane&15,
    // K = (lane>>4)*16 + e  -> 16 contiguous f16 from hT row N.
    const int nB = lane & 15;
    const int kB = (lane >> 4) * 16;
    const _Float16* hb = hT + (size_t)nB * N_NODES + (wave * 32) + kB;

    // Constant trip count (8192 / (4 waves * 32)) -> uniform scalar loop.
#pragma unroll 1
    for (int it = 0; it < N_NODES / 128; ++it) {
        // stream adj without polluting L2 (hT/s2 should stay resident)
        f4 av0 = __builtin_nontemporal_load((const f4*)(ap));
        f4 av1 = __builtin_nontemporal_load((const f4*)(ap + 4));
        f4 av2 = __builtin_nontemporal_load((const f4*)(ap + 8));
        f4 av3 = __builtin_nontemporal_load((const f4*)(ap + 12));
        __builtin_prefetch(ap + 128, 0, 3);   // next chunk for this wave

        f4 sv0 = *(const f4*)(sp);
        f4 sv1 = *(const f4*)(sp + 4);
        f4 sv2 = *(const f4*)(sp + 8);
        f4 sv3 = *(const f4*)(sp + 12);

        float av[16], sv[16];
#pragma unroll
        for (int q = 0; q < 4; ++q) {
            av[q]      = av0[q];  sv[q]      = sv0[q];
            av[4 + q]  = av1[q];  sv[4 + q]  = sv1[q];
            av[8 + q]  = av2[q];  sv[8 + q]  = sv2[q];
            av[12 + q] = av3[q];  sv[12 + q] = sv3[q];
        }

        v16h A;
        float prs[4] = {0.0f, 0.0f, 0.0f, 0.0f};
#pragma unroll
        for (int e = 0; e < 16; ++e) {
            float s  = s1v + sv[e];
            float lr = (s >= 0.0f) ? s : 0.2f * s;          // leaky_relu(0.2)
            float ex = __expf(lr);                           // unconditional
            float p  = (av[e] > 0.0f) ? ex : 0.0f;           // mask (cndmask)
            prs[e & 3] += p;                                 // 4-way sum tree
            A[e] = (_Float16)p;
        }
        rs0 += prs[0]; rs1 += prs[1]; rs2 += prs[2]; rs3 += prs[3];

        v16h B0 = *(const v16h*)(hb);
        v16h B1 = *(const v16h*)(hb + (size_t)16 * N_NODES);
        v16h B2 = *(const v16h*)(hb + (size_t)32 * N_NODES);
        v16h B3 = *(const v16h*)(hb + (size_t)48 * N_NODES);

        c0 = __builtin_amdgcn_wmma_f32_16x16x32_f16(false, A, false, B0, (short)0, c0, false, false);
        c1 = __builtin_amdgcn_wmma_f32_16x16x32_f16(false, A, false, B1, (short)0, c1, false, false);
        c2 = __builtin_amdgcn_wmma_f32_16x16x32_f16(false, A, false, B2, (short)0, c2, false, false);
        c3 = __builtin_amdgcn_wmma_f32_16x16x32_f16(false, A, false, B3, (short)0, c3, false, false);

        ap += 128;   // this wave's next K chunk
        sp += 128;
        hb += 128;
    }

    // Spill per-wave partials to LDS.
    rbuf[wave][lane] = rs0 + rs1 + rs2 + rs3;
    // C layout (f32 16x16): VGPR r, lanes 0-15: M=r,N=lane; lanes 16-31: M=r+8.
    const int crow = (lane >> 4) * 8;
    const int ccol = lane & 15;
#pragma unroll
    for (int r = 0; r < 8; ++r) {
        cbuf[wave][crow + r][ccol]      = c0[r];
        cbuf[wave][crow + r][16 + ccol] = c1[r];
        cbuf[wave][crow + r][32 + ccol] = c2[r];
        cbuf[wave][crow + r][48 + ccol] = c3[r];
    }
    __syncthreads();

    // Cross-wave reduction, softmax normalization, ELU, store.
    for (int idx = threadIdx.x; idx < 16 * 64; idx += 128) {
        int row = idx >> 6;
        int col = idx & 63;
        float acc = cbuf[0][row][col] + cbuf[1][row][col]
                  + cbuf[2][row][col] + cbuf[3][row][col];
        float den = 0.0f;
#pragma unroll
        for (int w = 0; w < 4; ++w)
            den += rbuf[w][row] + rbuf[w][row + 16];
        float v = acc / den;
        out[(size_t)(row0 + row) * F_OUT + col] =
            (v > 0.0f) ? v : (__expf(v) - 1.0f);            // elu, alpha=1
    }
}

// ---------------------------------------------------------------------------
extern "C" void kernel_launch(void* const* d_in, const int* in_sizes, int n_in,
                              void* d_out, int out_size, void* d_ws, size_t ws_size,
                              hipStream_t stream) {
    const float* x   = (const float*)d_in[0];   // 8192 x 128
    const float* adj = (const float*)d_in[1];   // 8192 x 8192
    const float* W   = (const float*)d_in[2];   // 128 x 64
    const float* a   = (const float*)d_in[3];   // 128
    float* out = (float*)d_out;                 // 8192 x 64

    char* ws = (char*)d_ws;
    _Float16* hT  = (_Float16*)ws;                       // 64*8192*2  = 1 MiB
    float*    h32 = (float*)(ws + (1u << 20));           // 8192*64*4  = 2 MiB
    float*    s1  = (float*)(ws + (3u << 20));           // 32 KiB
    float*    s2  = (float*)(ws + (3u << 20) + 32768u);  // 32 KiB

    gat_h_kernel<<<(N_NODES * F_OUT) / 256, 256, 0, stream>>>(x, W, h32, hT);
    gat_s_kernel<<<N_NODES / 256, 256, 0, stream>>>(h32, a, s1, s2);
    gat_main_kernel<<<N_NODES / 16, 128, 0, stream>>>(adj, s1, s2, hT, out);
}
